// DeformConv_7344394076343
// MI455X (gfx1250) — compile-verified, matrix operand order checked by reading, not run on hardware
//
#include <hip/hip_runtime.h>
#include <hip/hip_bf16.h>

typedef __attribute__((ext_vector_type(16))) _Float16 v16h;
typedef __attribute__((ext_vector_type(8)))  _Float16 v8h;
typedef __attribute__((ext_vector_type(4)))  _Float16 v4h;
typedef __attribute__((ext_vector_type(8)))  float    v8f;
typedef __attribute__((ext_vector_type(4)))  float    v4f;
typedef __attribute__((ext_vector_type(4)))  unsigned int v4u;
typedef __attribute__((ext_vector_type(8)))  int      v8i;
typedef __attribute__((ext_vector_type(4)))  int      v4i;

// Problem constants (match reference)
#define BATCH 8
#define CIN   256
#define HH    64
#define WW    64
#define HWSZ  4096          // H*W
#define OUTC  256
#define KTAPS 9             // 3x3
#define KTOT  2304          // CIN*KTAPS
#define NTILE 64            // pixels per workgroup (one output row)
#define KSTEP 32            // WMMA K per iteration
#define NKSTEPS 72          // KTOT / KSTEP
#define LDA   40            // padded row stride in halves (80B = 64B data + 16B TDM pad)
#define NSAMP 576           // KTAPS * NTILE bilinear samples per workgroup

// ---------------- prep: weights fp32 -> f16 into workspace ----------------
__global__ __launch_bounds__(256, 4)
void cvt_weights_f16(const float* __restrict__ wgt, _Float16* __restrict__ wh) {
    int i = (blockIdx.x * 256 + threadIdx.x) * 4;   // 589824 total, 4 per thread
    v4f w = *(const v4f*)(wgt + i);
    v4h h;
    h.x = (_Float16)w.x; h.y = (_Float16)w.y;
    h.z = (_Float16)w.z; h.w = (_Float16)w.w;
    *(v4h*)(wh + i) = h;
}

// Load a 16x32 f16 WMMA A/B fragment from an LDS row laid out K-contiguous.
// Lane layout (ISA 7.12.2): half = lane>>4 selects K blocks {8h..8h+7, 16+8h..16+8h+7}.
__device__ __forceinline__ v16h load_frag(const _Float16* row, int hh) {
    v8h c0 = *(const v8h*)(row + hh);        // K = 8*half .. 8*half+7
    v8h c1 = *(const v8h*)(row + 16 + hh);   // K = 16+8*half .. 16+8*half+7
    return __builtin_shufflevector(c0, c1, 0,1,2,3,4,5,6,7,8,9,10,11,12,13,14,15);
}

// Issue a TDM load of the 256x32 f16 weight tile into sA (wave-level DMA).
// TDM pad feature inserts 16B after every 64B row -> 80B LDS row stride (LDA=40).
__device__ __forceinline__ void tdm_load_weights(unsigned int lds_addr,
                                                 unsigned long long gaddr) {
    v4u g0;
    g0.x = 1u;                                   // count=1, user descriptor
    g0.y = lds_addr;                             // lds_addr[31:0] (bytes)
    g0.z = (unsigned int)(gaddr & 0xFFFFFFFFull);        // global_addr[31:0]
    g0.w = (unsigned int)((gaddr >> 32) & 0x1FFFFFFull)  // global_addr[56:32]
         | 0x80000000u;                          // type=2 ("image") in bits 127:126
    v8i g1;
    g1[0] = (int)((1u << 16)                     // data_size = 1 -> 2 bytes
                | (1u << 20)                     // pad_enable
                | (3u << 22)                     // pad_interval: 16 DWORDs = 64B
                | (3u << 25));                   // pad_amount:    4 DWORDs = 16B
    g1[1] = (int)(4096u << 16);                  // tensor_dim0[15:0]=4096 in [31:16]
    g1[2] = (int)(256u << 16);                   // tensor_dim1[15:0]=256  in [31:16]
    g1[3] = (int)(32u << 16);                    // tile_dim0=32 in [31:16]
    g1[4] = 256;                                 // tile_dim1=256
    g1[5] = KTOT;                                // tensor_dim0_stride = 2304 elements
    g1[6] = 0;
    g1[7] = 0;
    v4i g2 = {};                                 // 2-D tensor: groups 2/3 unused
    v4i g3 = {};
#if __clang_major__ >= 23
    v8i gx = {};
    __builtin_amdgcn_tensor_load_to_lds(g0, g1, g2, g3, gx, 0);
#else
    __builtin_amdgcn_tensor_load_to_lds(g0, g1, g2, g3, 0);
#endif
}

__global__ __launch_bounds__(256, 1)
void deform_conv_wmma(const float* __restrict__ x,
                      const float* __restrict__ off,
                      const _Float16* __restrict__ wh,
                      float* __restrict__ out) {
    __shared__ __align__(16) _Float16 sA[OUTC * LDA];   // 20480 B (TDM destination)
    __shared__ __align__(16) _Float16 sB[NTILE * LDA];  //  5120 B
    __shared__ float sW[NSAMP * 4];                     //  9216 B  corner weights
    __shared__ int   sI[NSAMP * 4];                     //  9216 B  corner indices

    const int tid   = threadIdx.x;
    const int lane  = tid & 31;
    const int wave  = tid >> 5;          // 0..7
    const int half  = lane >> 4;         // 0/1
    const int hh    = half << 3;         // 8*half
    const int ln16  = lane & 15;

    const int b     = blockIdx.y;        // batch
    const int pBase = blockIdx.x * NTILE;

    const unsigned int sA_lds = (unsigned int)(unsigned long long)(&sA[0]);

    // ---- Stage 0: per-workgroup bilinear sample descriptors (offset-dependent only) ----
    for (int s = tid; s < NSAMP; s += 256) {
        int kk = s >> 6;                 // tap 0..8
        int n  = s & 63;                 // pixel in tile
        int p  = pBase + n;
        int ho = p >> 6, wo = p & 63;
        int ki = kk / 3, kj = kk - 3 * (kk / 3);
        float sy = (float)(ho - 1 + ki) + off[(size_t)((b * 18 + 2 * kk    ) << 12) + p];
        float sx = (float)(wo - 1 + kj) + off[(size_t)((b * 18 + 2 * kk + 1) << 12) + p];
        float y0f = floorf(sy), x0f = floorf(sx);
        int   y0  = (int)y0f,   x0  = (int)x0f;
        float ly = sy - y0f, lx = sx - x0f;
        #pragma unroll
        for (int cy = 0; cy < 2; ++cy) {
            #pragma unroll
            for (int cx = 0; cx < 2; ++cx) {
                int yy = y0 + cy, xx = x0 + cx;
                bool valid = ((unsigned)yy < (unsigned)HH) && ((unsigned)xx < (unsigned)WW);
                float wy = cy ? ly : (1.0f - ly);
                float wx = cx ? lx : (1.0f - lx);
                int yc = yy < 0 ? 0 : (yy > HH - 1 ? HH - 1 : yy);
                int xc = xx < 0 ? 0 : (xx > WW - 1 ? WW - 1 : xx);
                sW[(s << 2) + (cy * 2 + cx)] = valid ? (wy * wx) : 0.0f;
                sI[(s << 2) + (cy * 2 + cx)] = yc * WW + xc;
            }
        }
    }
    __syncthreads();

    // f32 accumulators: wave covers m-rows [32w, 32w+32) x all 64 pixels -> 2x4 WMMA tiles
    v8f acc[2][4];
    #pragma unroll
    for (int mi = 0; mi < 2; ++mi)
        #pragma unroll
        for (int j = 0; j < 4; ++j)
            acc[mi][j] = v8f{};

    const float* xb_base = x + ((size_t)(b << 8) << 12);   // x[b][0][0][0]

    for (int ks = 0; ks < NKSTEPS; ++ks) {
        // ---- Stage A: TDM DMA of the 256x32 f16 weight tile, overlapped with Stage B.
        // Wave-uniform branch; TDM ignores EXEC and is issued once by wave 0.
        if (wave == 0) {
            unsigned long long ga =
                (unsigned long long)(const void*)wh + (unsigned long long)(ks * KSTEP * 2);
            tdm_load_weights(sA_lds, ga);
        }

        // ---- Stage B: im2col tile 32(K) x 64(N), bilinear gather, store N-major f16 ----
        #pragma unroll
        for (int i = 0; i < 8; ++i) {
            int e  = tid + (i << 8);          // 0..2047
            int t  = e & 31;                  // k within step
            int n  = e >> 5;                  // pixel in tile
            int r  = ks * KSTEP + t;          // global reduction index = c*9 + kk
            int c  = r / KTAPS;
            int kk = r - c * KTAPS;
            int s  = (kk << 6) + n;
            const float* w4 = &sW[s << 2];
            const int*   i4 = &sI[s << 2];
            const float* xc = xb_base + ((size_t)c << 12);
            float v = w4[0] * xc[i4[0]] + w4[1] * xc[i4[1]]
                    + w4[2] * xc[i4[2]] + w4[3] * xc[i4[3]];
            sB[n * LDA + t] = (_Float16)v;
        }

        // Wave 0 fences its TDM transfer; barrier publishes sA + sB to all waves.
        if (wave == 0)
            __builtin_amdgcn_s_wait_tensorcnt(0);
        __syncthreads();

        // ---- Compute: 8 x v_wmma_f32_16x16x32_f16 per wave ----
        v16h afr[2], bfr[4];
        #pragma unroll
        for (int mi = 0; mi < 2; ++mi)
            afr[mi] = load_frag(&sA[((wave << 5) + (mi << 4) + ln16) * LDA], hh);
        #pragma unroll
        for (int j = 0; j < 4; ++j)
            bfr[j] = load_frag(&sB[((j << 4) + ln16) * LDA], hh);

        #pragma unroll
        for (int mi = 0; mi < 2; ++mi)
            #pragma unroll
            for (int j = 0; j < 4; ++j)
                acc[mi][j] = __builtin_amdgcn_wmma_f32_16x16x32_f16(
                    false, afr[mi], false, bfr[j],
                    (short)0, acc[mi][j], false, false);

        __syncthreads();   // protect LDS tiles before next iteration's DMA/gather
    }

    // ---- Epilogue: C/D layout -> out[b][o][p] ----
    // D frag: lane n = ln16 (+16*j), VGPR v -> m = v + 8*half (+ frag base)
    const size_t outb = ((size_t)b << 8) << 12;   // b * OUTC * HWSZ
    #pragma unroll
    for (int mi = 0; mi < 2; ++mi) {
        #pragma unroll
        for (int j = 0; j < 4; ++j) {
            int ng = pBase + (j << 4) + ln16;
            #pragma unroll
            for (int v = 0; v < 8; ++v) {
                int o = (wave << 5) + (mi << 4) + v + hh;
                out[outb + ((size_t)o << 12) + ng] = acc[mi][j][v];
            }
        }
    }
}

extern "C" void kernel_launch(void* const* d_in, const int* in_sizes, int n_in,
                              void* d_out, int out_size, void* d_ws, size_t ws_size,
                              hipStream_t stream) {
    const float* x   = (const float*)d_in[0];   // [8,256,64,64]
    const float* off = (const float*)d_in[1];   // [8,18,64,64]
    const float* wgt = (const float*)d_in[2];   // [256,256,3,3]
    float* out = (float*)d_out;                 // [8,256,64,64]
    _Float16* wh = (_Float16*)d_ws;             // f16 weights [256][2304], 1.18 MB

    // 1) convert weights to f16 once (589824 elems, 4/thread)
    cvt_weights_f16<<<dim3(OUTC * KTOT / (256 * 4)), dim3(256), 0, stream>>>(wgt, wh);

    // 2) fused im2col + WMMA GEMM, weight tiles streamed by the Tensor Data Mover
    dim3 grid(HWSZ / NTILE, BATCH, 1);          // 64 pixel-tiles x 8 batches
    dim3 block(256, 1, 1);                      // 8 waves (wave32)
    deform_conv_wmma<<<grid, block, 0, stream>>>(x, off, wh, out);
}
